// TransformerWithWindowAttention_88218628260829
// MI455X (gfx1250) — matrix operand order repaired and checked by reading, not compiled, not run on hardware
//
#include <hip/hip_runtime.h>

// ---------------------------------------------------------------------------
// Types for CDNA5 WMMA (wave32).
// ---------------------------------------------------------------------------
typedef __attribute__((ext_vector_type(16))) __bf16 v16bf;
typedef __attribute__((ext_vector_type(8)))  float  v8f;

#define CEMB 512      // embedding dim
#define NHEAD 8
#define HD 64         // head dim
#define WIN 64        // window size
#define LDSS 40       // LDS row stride (32 K + 8 pad) for GEMM tiles
#define PSTR 72       // LDS row stride for 64-wide attention tiles

static __device__ __forceinline__ float gelu_exact(float x) {
    return 0.5f * x * (1.0f + erff(x * 0.70710678118654752f));
}
static __device__ __forceinline__ __bf16 tobf(float x) { return (__bf16)x; }

// ---------------------------------------------------------------------------
// Embed: X[t,c] = gelu(x[t]*fe_w[c] + fe_b[c]) + pos[t%L0, c]
// ---------------------------------------------------------------------------
__global__ void embed_kernel(const float* __restrict__ x,
                             const float* __restrict__ fe_w,
                             const float* __restrict__ fe_b,
                             const float* __restrict__ pos,
                             float* __restrict__ X,
                             int L0, long long total) {
    long long idx = (long long)blockIdx.x * blockDim.x + threadIdx.x;
    if (idx >= total) return;
    int c = (int)(idx & (CEMB - 1));
    long long t = idx >> 9;
    int i = (int)(t % L0);
    float v = x[t];                       // x is (B,1,L0) flat == token order
    float h = gelu_exact(v * fe_w[c] + fe_b[c]);
    X[idx] = h + pos[(long long)i * CEMB + c];
}

// ---------------------------------------------------------------------------
// LayerNorm over C=512 with optional cyclic roll (shifted windows).
// out[b, i, :] = LN(in[b, (i+shift)%L, :]) * g + bta
// One 128-thread block per output row; 4 elements per thread.
// ---------------------------------------------------------------------------
__global__ __launch_bounds__(128)
void ln_kernel(const float* __restrict__ in, float* __restrict__ out,
               const float* __restrict__ g, const float* __restrict__ bta,
               int L, int shift) {
    int row = blockIdx.x;
    int b = row / L;
    int i = row - b * L;
    int src = b * L + (i + shift) % L;
    const float* xp = in + (long long)src * CEMB;
    int t = threadIdx.x;

    float v0 = xp[t], v1 = xp[t + 128], v2 = xp[t + 256], v3 = xp[t + 384];
    float s = v0 + v1 + v2 + v3;
    float q = v0 * v0 + v1 * v1 + v2 * v2 + v3 * v3;
    #pragma unroll
    for (int off = 1; off < 32; off <<= 1) {
        s += __shfl_xor(s, off, 32);
        q += __shfl_xor(q, off, 32);
    }
    __shared__ float ss[4], qq[4];
    if ((t & 31) == 0) { ss[t >> 5] = s; qq[t >> 5] = q; }
    __syncthreads();
    s = ss[0] + ss[1] + ss[2] + ss[3];
    q = qq[0] + qq[1] + qq[2] + qq[3];
    float mean = s * (1.0f / CEMB);
    float var  = q * (1.0f / CEMB) - mean * mean;
    float inv  = rsqrtf(var + 1e-5f);

    float* op = out + (long long)row * CEMB;
    op[t]       = (v0 - mean) * inv * g[t]       + bta[t];
    op[t + 128] = (v1 - mean) * inv * g[t + 128] + bta[t + 128];
    op[t + 256] = (v2 - mean) * inv * g[t + 256] + bta[t + 256];
    op[t + 384] = (v3 - mean) * inv * g[t + 384] + bta[t + 384];
}

// ---------------------------------------------------------------------------
// Tiled bf16 WMMA GEMM:  out[M,N] = act(A[M,K] @ W[N,K]^T + bias) (+ resid)
// 256 threads = 8 waves. Block tile 64x64; wave -> 16x32 strip (2 accums).
// A and W staged f32->bf16 in LDS each K-step of 32.
// ---------------------------------------------------------------------------
__global__ __launch_bounds__(256)
void gemm_bias_kernel(const float* __restrict__ A, const float* __restrict__ W,
                      const float* __restrict__ bias,
                      const float* __restrict__ resid,
                      float* __restrict__ out,
                      int M, int N, int K, int do_gelu) {
    __shared__ __bf16 sA[64 * LDSS];
    __shared__ __bf16 sB[64 * LDSS];

    const int tid  = threadIdx.x;
    const int bm   = blockIdx.y * 64;
    const int bn   = blockIdx.x * 64;
    const int wave = tid >> 5;
    const int lane = tid & 31;
    const int half = lane >> 4;
    const int l16  = lane & 15;
    const int wrt  = wave & 3;        // wave row tile (16 rows)
    const int wct  = wave >> 2;       // wave col group (32 cols)

    const int lr = tid >> 2;          // cooperative-load row 0..63
    const int lk = (tid & 3) * 8;     // cooperative-load k segment

    v8f acc0 = {};
    v8f acc1 = {};

    for (int k0 = 0; k0 < K; k0 += 32) {
        const float* ap = A + (long long)(bm + lr) * K + k0 + lk;
        const float* bp = W + (long long)(bn + lr) * K + k0 + lk;
        if (k0 + 32 < K) {                       // prefetch next K tile
            __builtin_prefetch(ap + 32, 0, 1);
            __builtin_prefetch(bp + 32, 0, 1);
        }
        #pragma unroll
        for (int j = 0; j < 8; ++j) {
            sA[lr * LDSS + lk + j] = tobf(ap[j]);
            sB[lr * LDSS + lk + j] = tobf(bp[j]);
        }
        __syncthreads();

        const __bf16* arow  = &sA[(16 * wrt + l16) * LDSS];
        const __bf16* brow0 = &sB[(32 * wct + l16) * LDSS];
        const __bf16* brow1 = brow0 + 16 * LDSS;

        v16bf af, bf0, bf1;
        #pragma unroll
        for (int j = 0; j < 8; ++j) {            // A: k = half*8+j, 16+half*8+j
            af[j]     = arow[half * 8 + j];
            af[j + 8] = arow[16 + half * 8 + j];
        }
        #pragma unroll
        for (int j = 0; j < 8; ++j) {            // B: k = half*8+j, 16+half*8+j
            bf0[j]     = brow0[half * 8 + j];
            bf0[j + 8] = brow0[16 + half * 8 + j];
            bf1[j]     = brow1[half * 8 + j];
            bf1[j + 8] = brow1[16 + half * 8 + j];
        }
        acc0 = __builtin_amdgcn_wmma_f32_16x16x32_bf16(false, af, false, bf0,
                                                       (short)0, acc0, false, false);
        acc1 = __builtin_amdgcn_wmma_f32_16x16x32_bf16(false, af, false, bf1,
                                                       (short)0, acc1, false, false);
        __syncthreads();
    }

    #pragma unroll
    for (int r = 0; r < 8; ++r) {
        int row = bm + 16 * wrt + 8 * half + r;
        int c0  = bn + 32 * wct + l16;
        int c1  = c0 + 16;
        float v0 = acc0[r] + bias[c0];
        float v1 = acc1[r] + bias[c1];
        if (do_gelu) { v0 = gelu_exact(v0); v1 = gelu_exact(v1); }
        long long o0 = (long long)row * N + c0;
        long long o1 = (long long)row * N + c1;
        if (resid) { v0 += resid[o0]; v1 += resid[o1]; }
        out[o0] = v0;
        out[o1] = v1;
    }
}

// ---------------------------------------------------------------------------
// Windowed attention. Block = (window, head), 128 threads = 4 waves.
// qkv layout per token: [q(512) | k(512) | v(512)], head slice h*64.
// Wave w owns query rows [16w, 16w+16): S strip (4 accums), softmax,
// P (bf16) -> LDS, then O strip = P @ V with V staged bf16 in LDS.
// ---------------------------------------------------------------------------
__global__ __launch_bounds__(128)
void window_attn_kernel(const float* __restrict__ qkv,
                        const float* __restrict__ bias_table,
                        float* __restrict__ out) {
    __shared__ __bf16 sP[64 * PSTR];
    __shared__ __bf16 sV[64 * PSTR];
    __shared__ float  sBias[2 * WIN - 1];

    const int win0 = blockIdx.x * WIN;     // first token of window
    const int h    = blockIdx.y;
    const int tid  = threadIdx.x;
    const int wave = tid >> 5;
    const int lane = tid & 31;
    const int half = lane >> 4;
    const int l16  = lane & 15;

    if (tid < 2 * WIN - 1) sBias[tid] = bias_table[tid * NHEAD + h];
    for (int idx = tid; idx < WIN * HD; idx += 128) {
        int k = idx >> 6, d = idx & 63;
        sV[k * PSTR + d] =
            tobf(qkv[(long long)(win0 + k) * (3 * CEMB) + 2 * CEMB + h * HD + d]);
    }
    __syncthreads();

    // ---- S = Q @ K^T -------------------------------------------------------
    v8f s0 = {}, s1 = {}, s2 = {}, s3 = {};
    #pragma unroll
    for (int kk = 0; kk < HD; kk += 32) {
        const float* qrow =
            qkv + (long long)(win0 + 16 * wave + l16) * (3 * CEMB) + h * HD + kk;
        v16bf af;
        #pragma unroll
        for (int j = 0; j < 8; ++j) {
            af[j]     = tobf(qrow[half * 8 + j]);
            af[j + 8] = tobf(qrow[16 + half * 8 + j]);
        }
        #pragma unroll
        for (int nt = 0; nt < 4; ++nt) {
            const float* krow =
                qkv + (long long)(win0 + 16 * nt + l16) * (3 * CEMB) + CEMB + h * HD + kk;
            v16bf bf;
            #pragma unroll
            for (int j = 0; j < 8; ++j) {
                bf[j]     = tobf(krow[half * 8 + j]);
                bf[j + 8] = tobf(krow[16 + half * 8 + j]);
            }
            v8f* acc = (nt == 0) ? &s0 : (nt == 1) ? &s1 : (nt == 2) ? &s2 : &s3;
            *acc = __builtin_amdgcn_wmma_f32_16x16x32_bf16(false, af, false, bf,
                                                           (short)0, *acc, false, false);
        }
    }

    // ---- scale + relative bias + softmax (rows m = 16w + 8*half + r) ------
    const float scale = 0.125f;   // hd^-0.5, hd = 64
    #pragma unroll
    for (int r = 0; r < 8; ++r) {
        int m = 16 * wave + 8 * half + r;
        float b0 = sBias[m - (0  + l16) + (WIN - 1)];
        float b1 = sBias[m - (16 + l16) + (WIN - 1)];
        float b2 = sBias[m - (32 + l16) + (WIN - 1)];
        float b3 = sBias[m - (48 + l16) + (WIN - 1)];
        float e0 = s0[r] * scale + b0;
        float e1 = s1[r] * scale + b1;
        float e2 = s2[r] * scale + b2;
        float e3 = s3[r] * scale + b3;
        float mx = fmaxf(fmaxf(e0, e1), fmaxf(e2, e3));
        #pragma unroll
        for (int off = 1; off < 16; off <<= 1) mx = fmaxf(mx, __shfl_xor(mx, off, 32));
        e0 = expf(e0 - mx); e1 = expf(e1 - mx);
        e2 = expf(e2 - mx); e3 = expf(e3 - mx);
        float su = e0 + e1 + e2 + e3;
        #pragma unroll
        for (int off = 1; off < 16; off <<= 1) su += __shfl_xor(su, off, 32);
        float inv = 1.0f / su;
        sP[m * PSTR + (0  + l16)] = tobf(e0 * inv);
        sP[m * PSTR + (16 + l16)] = tobf(e1 * inv);
        sP[m * PSTR + (32 + l16)] = tobf(e2 * inv);
        sP[m * PSTR + (48 + l16)] = tobf(e3 * inv);
    }
    __syncthreads();

    // ---- O = P @ V ---------------------------------------------------------
    v8f o0 = {}, o1 = {}, o2 = {}, o3 = {};
    #pragma unroll
    for (int kk = 0; kk < WIN; kk += 32) {
        const __bf16* prow = &sP[(16 * wave + l16) * PSTR + kk];
        v16bf af;
        #pragma unroll
        for (int j = 0; j < 8; ++j) {
            af[j]     = prow[half * 8 + j];
            af[j + 8] = prow[16 + half * 8 + j];
        }
        #pragma unroll
        for (int nt = 0; nt < 4; ++nt) {
            int n = 16 * nt + l16;
            v16bf bf;
            #pragma unroll
            for (int j = 0; j < 8; ++j) {
                bf[j]     = sV[(kk + half * 8 + j) * PSTR + n];
                bf[j + 8] = sV[(kk + 16 + half * 8 + j) * PSTR + n];
            }
            v8f* acc = (nt == 0) ? &o0 : (nt == 1) ? &o1 : (nt == 2) ? &o2 : &o3;
            *acc = __builtin_amdgcn_wmma_f32_16x16x32_bf16(false, af, false, bf,
                                                           (short)0, *acc, false, false);
        }
    }

    #pragma unroll
    for (int r = 0; r < 8; ++r) {
        int m = 16 * wave + 8 * half + r;
        long long base = (long long)(win0 + m) * CEMB + h * HD;
        out[base + 0  + l16] = o0[r];
        out[base + 16 + l16] = o1[r];
        out[base + 32 + l16] = o2[r];
        out[base + 48 + l16] = o3[r];
    }
}

// ---------------------------------------------------------------------------
// X[b,i,:] += mask * P[b,(i-shift)%L,:]  (inverse roll + boundary mask)
// ---------------------------------------------------------------------------
__global__ void add_roll_mask_kernel(float* __restrict__ X,
                                     const float* __restrict__ P,
                                     int shift, int L, long long total) {
    long long idx = (long long)blockIdx.x * blockDim.x + threadIdx.x;
    if (idx >= total) return;
    int c = (int)(idx & (CEMB - 1));
    long long tok = idx >> 9;
    int b = (int)(tok / L);
    int i = (int)(tok - (long long)b * L);
    int src = (i - shift + L) % L;
    float v = (src < shift || src >= L - shift)
                  ? 0.0f
                  : P[(((long long)b * L + src) << 9) + c];
    X[idx] += v;
}

// ---------------------------------------------------------------------------
// Patch-merge gather: G[b,j,2c+s] = X[b,2j+s,c]   (K = 1024)
// ---------------------------------------------------------------------------
__global__ void merge_gather_kernel(const float* __restrict__ X,
                                    float* __restrict__ G,
                                    int L2, long long total) {
    long long idx = (long long)blockIdx.x * blockDim.x + threadIdx.x;
    if (idx >= total) return;
    int k = (int)(idx & 1023);
    long long j = idx >> 10;
    int b  = (int)(j / L2);
    int jj = (int)(j - (long long)b * L2);
    int c = k >> 1, s2 = k & 1;
    G[idx] = X[(((long long)b * (2 * L2) + 2 * jj + s2) << 9) + c];
}

// ---------------------------------------------------------------------------
// Mean over sequence: red[b,c] = mean_i X[b,i,c]
// ---------------------------------------------------------------------------
__global__ void mean_kernel(const float* __restrict__ X, float* __restrict__ red,
                            int Bn, int L) {
    int idx = blockIdx.x * blockDim.x + threadIdx.x;
    if (idx >= Bn * CEMB) return;
    int b = idx >> 9, c = idx & (CEMB - 1);
    const float* p = X + (((long long)b * L) << 9) + c;
    float s = 0.0f;
    for (int i = 0; i < L; ++i) s += p[(long long)i << 9];
    red[idx] = s / (float)L;
}

// ---------------------------------------------------------------------------
// Classifier: out[b,o] = dot(h[b], w[o]) + bias[o]   (B*2 = 32 outputs)
// ---------------------------------------------------------------------------
__global__ void cls_kernel(const float* __restrict__ h, const float* __restrict__ w,
                           const float* __restrict__ bias, float* __restrict__ out,
                           int Bn) {
    int t = blockIdx.x * blockDim.x + threadIdx.x;
    if (t >= Bn * 2) return;
    int b = t >> 1, o = t & 1;
    float s = 0.0f;
    for (int c = 0; c < CEMB; ++c) s += h[b * CEMB + c] * w[o * CEMB + c];
    out[t] = s + bias[o];
}

// ---------------------------------------------------------------------------
// Host orchestration
// ---------------------------------------------------------------------------
static inline int cdiv_ll(long long a, int b) { return (int)((a + b - 1) / b); }

extern "C" void kernel_launch(void* const* d_in, const int* in_sizes, int n_in,
                              void* d_out, int out_size, void* d_ws, size_t ws_size,
                              hipStream_t stream) {
    (void)in_sizes; (void)n_in; (void)out_size; (void)ws_size;

    const int Bn = 16, L0 = 4096;
    const float* x_in = (const float*)d_in[0];
    const float* fe_w = (const float*)d_in[1];
    const float* fe_b = (const float*)d_in[2];
    const float* pos  = (const float*)d_in[3];
    // blocks: base 4 + blk*13
    //   +0 n1_w +1 n1_b +2 bias_table +3 qkv_w +4 qkv_b +5 proj_w +6 proj_b
    //   +7 n2_w +8 n2_b +9 fc1_w +10 fc1_b +11 fc2_w +12 fc2_b
    // pm: base 82 + i*4: conv_w conv_b norm_w norm_b ; cls: 90..93
    auto P_ = [&](int i) { return (const float*)d_in[i]; };

    float* ws = (float*)d_ws;
    const long long TOK0 = (long long)Bn * L0;          // 65536
    float* X    = ws;                                   // TOK0*512
    float* Abuf = X    + TOK0 * CEMB;                   // TOK0*512
    float* Obuf = Abuf + TOK0 * CEMB;                   // TOK0*512
    float* Pbuf = Obuf + TOK0 * CEMB;                   // TOK0*512
    float* QKV  = Pbuf + TOK0 * CEMB;                   // TOK0*1536 (also merge gather)
    float* M4   = QKV  + TOK0 * 3 * CEMB;               // TOK0*2048
    float* RED  = M4   + TOK0 * 4 * CEMB;               // Bn*512
    float* RED2 = RED  + Bn * CEMB;                     // Bn*512

    // ---- embed + pos ----
    {
        long long total = TOK0 * CEMB;
        embed_kernel<<<cdiv_ll(total, 256), 256, 0, stream>>>(
            x_in, fe_w, fe_b, pos, X, L0, total);
    }

    int L = L0;
    for (int stage = 0; stage < 3; ++stage) {
        long long tokens = (long long)Bn * L;
        long long totC   = tokens * CEMB;
        for (int s = 0; s < 2; ++s) {
            int blk   = stage * 2 + s;
            int shift = (s == 0) ? 0 : WIN / 2;
            int base  = 4 + blk * 13;

            // LN1 with cyclic roll
            ln_kernel<<<(int)tokens, 128, 0, stream>>>(X, Abuf, P_(base + 0),
                                                       P_(base + 1), L, shift);
            // qkv = A @ qkv_w^T + b   (N=1536, K=512)
            gemm_bias_kernel<<<dim3(1536 / 64, (int)(tokens / 64)), 256, 0, stream>>>(
                Abuf, P_(base + 3), P_(base + 4), nullptr, QKV,
                (int)tokens, 1536, CEMB, 0);
            // windowed attention
            window_attn_kernel<<<dim3((int)(tokens / WIN), NHEAD), 128, 0, stream>>>(
                QKV, P_(base + 2), Obuf);
            // proj (N=512, K=512)
            gemm_bias_kernel<<<dim3(CEMB / 64, (int)(tokens / 64)), 256, 0, stream>>>(
                Obuf, P_(base + 5), P_(base + 6), nullptr, Pbuf,
                (int)tokens, CEMB, CEMB, 0);
            // residual with inverse roll + boundary mask
            add_roll_mask_kernel<<<cdiv_ll(totC, 256), 256, 0, stream>>>(
                X, Pbuf, shift, L, totC);
            // LN2
            ln_kernel<<<(int)tokens, 128, 0, stream>>>(X, Abuf, P_(base + 7),
                                                       P_(base + 8), L, 0);
            // fc1 + GELU (N=2048, K=512)
            gemm_bias_kernel<<<dim3(2048 / 64, (int)(tokens / 64)), 256, 0, stream>>>(
                Abuf, P_(base + 9), P_(base + 10), nullptr, M4,
                (int)tokens, 4 * CEMB, CEMB, 1);
            // fc2 + residual (N=512, K=2048)
            gemm_bias_kernel<<<dim3(CEMB / 64, (int)(tokens / 64)), 256, 0, stream>>>(
                M4, P_(base + 11), P_(base + 12), X, X,
                (int)tokens, CEMB, 4 * CEMB, 0);
        }
        if (stage < 2) {
            int pmb = 82 + stage * 4;
            int L2 = L / 2;
            long long tokens2 = (long long)Bn * L2;
            long long totG = tokens2 * 2 * CEMB;
            merge_gather_kernel<<<cdiv_ll(totG, 256), 256, 0, stream>>>(
                X, QKV, L2, totG);
            gemm_bias_kernel<<<dim3(CEMB / 64, (int)(tokens2 / 64)), 256, 0, stream>>>(
                QKV, P_(pmb + 0), P_(pmb + 1), nullptr, Obuf,
                (int)tokens2, CEMB, 2 * CEMB, 0);
            ln_kernel<<<(int)tokens2, 128, 0, stream>>>(Obuf, X, P_(pmb + 2),
                                                        P_(pmb + 3), L2, 0);
            L = L2;
        }
    }

    // ---- mean pool, final LN, classifier ----
    mean_kernel<<<cdiv_ll((long long)Bn * CEMB, 256), 256, 0, stream>>>(X, RED, Bn, L);
    ln_kernel<<<Bn, 128, 0, stream>>>(RED, RED2, P_(90), P_(91), 1, 0);
    cls_kernel<<<1, 32, 0, stream>>>(RED2, P_(92), P_(93), (float*)d_out, Bn);
}
